// InterpretableOTSNAKOrNLayer_22694607192465
// MI455X (gfx1250) — matrix operand order repaired and checked by reading, not compiled, not run on hardware
//
#include <hip/hip_runtime.h>
#include <hip/hip_bf16.h>

typedef _Float16 f16;
typedef __attribute__((ext_vector_type(16))) _Float16 v16h;
typedef __attribute__((ext_vector_type(8)))  float    v8f;

#define DT_C 0.014f

__device__ __forceinline__ v8f zero8() {
  v8f z;
  for (int i = 0; i < 8; ++i) z[i] = 0.0f;
  return z;
}

// Pack two f32 into two f16 in one dword (v_cvt_pk_f16_f32).
__device__ __forceinline__ unsigned pk2(float a, float b) {
  union { unsigned u; f16 h[2]; } p;
  p.h[0] = (f16)a;
  p.h[1] = (f16)b;
  return p.u;
}

// Load a 32-byte WMMA fragment (16 x f16) from two 16-byte chunks.
__device__ __forceinline__ v16h ld_frag2(const f16* p0, const f16* p1) {
  union { v16h v; uint4 u[2]; } r;
  r.u[0] = *(const uint4*)p0;
  r.u[1] = *(const uint4*)p1;
  return r.v;
}

// ---------------------------------------------------------------------------
// Kernel 1: pack J (512x512 f32) into f16 A-fragment layout.
// Tile (mt,kt) is 16(M) x 32(K). Lane L holds:
//   row = mt*16 + (L&15), hi = L>>4
//   bytes[ 0:15] = J[row, kt*32 + 8*hi + 0..7]
//   bytes[16:31] = J[row, kt*32 + 16 + 8*hi + 0..7]
// ---------------------------------------------------------------------------
__global__ __launch_bounds__(256) void packJ_kernel(const float* __restrict__ J,
                                                    f16* __restrict__ Jp) {
  const int tid = threadIdx.x;
  const int gw  = blockIdx.x * 8 + (tid >> 5);  // tile id 0..511
  const int ln  = tid & 31;
  const int mt  = gw >> 4;
  const int kt  = gw & 15;
  const int row = mt * 16 + (ln & 15);
  const int hi  = ln >> 4;
  const float* src = J + row * 512 + kt * 32;
  union { uint4 u[2]; f16 h[16]; } t;
  for (int i = 0; i < 8; ++i) t.h[i]     = (f16)src[hi * 8 + i];
  for (int i = 0; i < 8; ++i) t.h[8 + i] = (f16)src[16 + hi * 8 + i];
  uint4* dst = (uint4*)(Jp + (gw * 32 + ln) * 16);
  dst[0] = t.u[0];
  dst[1] = t.u[1];
}

// ---------------------------------------------------------------------------
// Kernel 2: fused  c = tanh( gelu(LN(X@W1 + b1)) @ W2 + b2 )  for 16 rows/WG.
// Only 4 accumulator fragments are live at a time: each 512-col q-block is
// immediately reduced into LN stats (f32, LDS atomics) and stored pre-LN to
// sH (f16); LN+GELU is then one in-place elementwise pass.
// ---------------------------------------------------------------------------
__global__ __launch_bounds__(256) void mlp_kernel(
    const float* __restrict__ X,    // features (256,256,512) f32
    const float* __restrict__ W1,   // 512 x 2048
    const float* __restrict__ b1,   // 2048
    const float* __restrict__ lng,  // 2048
    const float* __restrict__ lnb,  // 2048
    const float* __restrict__ W2,   // 2048 x 1024
    const float* __restrict__ b2,   // 1024
    f16* __restrict__ Cout)         // 32768 x 1024 (f16)
{
  __shared__ __align__(16) f16 sA[16 * 520];    // A block, row stride 520
  __shared__ __align__(16) f16 sW[512 * 40];    // W slab, col-major [n][k], stride 40
  __shared__ __align__(16) f16 sH[16 * 2056];   // h block, row stride 2056
  __shared__ float sSum[16];
  __shared__ float sSsq[16];
  __shared__ float sMu[16];
  __shared__ float sRstd[16];

  const int tid = threadIdx.x;
  const int wv  = tid >> 5;
  const int ln  = tid & 31;
  const int lc  = ln & 15;
  const int hi  = ln >> 4;
  const int rowBase = blockIdx.x * 16;

  // ---- Load A block: 16 rows x 512, f32 -> packed f16 ----
  for (int m = 0; m < 16; ++m) {
    int idx = tid + (m << 8);       // 0..4095 (pairs)
    int i = idx >> 8;               // row 0..15
    int col = (idx & 255) * 2;
    int r = rowBase + i;
    int b = r >> 7;
    int s = r & 127;
    const float2 v = *(const float2*)&X[((b << 8) + (s << 1)) * 512 + col];
    *(unsigned*)&sA[i * 520 + col] = pk2(v.x, v.y);
  }
  if (tid < 16) { sSum[tid] = 0.0f; sSsq[tid] = 0.0f; }

  // ---- Phase 1: q-blocks of 512 columns; 4 live accumulators ----
  for (int q = 0; q < 4; ++q) {
    v8f acc[4];
    for (int f = 0; f < 4; ++f) acc[f] = zero8();

    for (int kt = 0; kt < 16; ++kt) {
      __syncthreads();
      // stage W1 slab: k in [kt*32,+32), n in [q*512,+512), packed pairs in k
      for (int m = 0; m < 32; ++m) {
        int idx = tid + (m << 8);   // 0..8191
        int n = idx & 511;
        int k = (idx >> 9) * 2;     // 0,2,..,30
        const float* wp = &W1[(kt * 32 + k) * 2048 + q * 512 + n];
        *(unsigned*)&sW[n * 40 + k] = pk2(wp[0], wp[2048]);
      }
      __syncthreads();
      const f16* ap = &sA[lc * 520 + kt * 32];
      v16h a = ld_frag2(ap + hi * 8, ap + 16 + hi * 8);
      v16h bf[4];
      for (int f = 0; f < 4; ++f) {
        const f16* bp = &sW[((wv * 4 + f) * 16 + lc) * 40 + hi * 8];
        bf[f] = ld_frag2(bp, bp + 16);
      }
      for (int f = 0; f < 4; ++f) {
        acc[f] = __builtin_amdgcn_wmma_f32_16x16x32_f16(
            false, a, false, bf[f], (short)0, acc[f], false, false);
      }
    }

    // q-epilogue: add b1, accumulate LN stats, store pre-LN f16 to sH
    float b1v[4];
    int nn[4];
    for (int f = 0; f < 4; ++f) {
      nn[f] = q * 512 + (wv * 4 + f) * 16 + lc;
      b1v[f] = b1[nn[f]];
    }
    for (int r = 0; r < 8; ++r) {
      float s = 0.0f, ss = 0.0f;
      for (int f = 0; f < 4; ++f) {
        float v = acc[f][r] + b1v[f];
        s += v; ss += v * v;
      }
      atomicAdd(&sSum[r + hi * 8], s);
      atomicAdd(&sSsq[r + hi * 8], ss);
    }
    for (int f = 0; f < 4; ++f)
      for (int r = 0; r < 8; ++r)
        sH[(r + hi * 8) * 2056 + nn[f]] = (f16)(acc[f][r] + b1v[f]);
  }

  __syncthreads();
  if (tid < 16) {
    float mean = sSum[tid] * (1.0f / 2048.0f);
    float var  = sSsq[tid] * (1.0f / 2048.0f) - mean * mean;
    sMu[tid] = mean;
    sRstd[tid] = rsqrtf(var + 1e-5f);
  }
  __syncthreads();

  // ---- In-place LayerNorm + exact GELU over sH ----
  for (int m = 0; m < 128; ++m) {
    int e = tid + (m << 8);         // 0..32767
    int row = e >> 11;
    int n = e & 2047;
    float v = (float)sH[row * 2056 + n];
    float h = (v - sMu[row]) * sRstd[row] * lng[n] + lnb[n];
    float ge = 0.5f * h * (1.0f + erff(h * 0.70710678118654752f));
    sH[row * 2056 + n] = (f16)ge;
  }

  // ---- Phase 2: c = tanh(h @ W2 + b2), K=2048, N=1024 in two 512 halves ----
  for (int half = 0; half < 2; ++half) {
    v8f a2[4];
    for (int f = 0; f < 4; ++f) a2[f] = zero8();
    for (int kt = 0; kt < 64; ++kt) {
      __syncthreads();
      for (int m = 0; m < 32; ++m) {
        int idx = tid + (m << 8);
        int n = idx & 511;
        int k = (idx >> 9) * 2;
        const float* wp = &W2[(kt * 32 + k) * 1024 + half * 512 + n];
        *(unsigned*)&sW[n * 40 + k] = pk2(wp[0], wp[1024]);
      }
      __syncthreads();
      const f16* ap = &sH[lc * 2056 + kt * 32];
      v16h a = ld_frag2(ap + hi * 8, ap + 16 + hi * 8);
      v16h bf[4];
      for (int f = 0; f < 4; ++f) {
        const f16* bp = &sW[((wv * 4 + f) * 16 + lc) * 40 + hi * 8];
        bf[f] = ld_frag2(bp, bp + 16);
      }
      for (int f = 0; f < 4; ++f) {
        a2[f] = __builtin_amdgcn_wmma_f32_16x16x32_f16(
            false, a, false, bf[f], (short)0, a2[f], false, false);
      }
    }
    for (int f = 0; f < 4; ++f) {
      int n = half * 512 + (wv * 4 + f) * 16 + lc;
      float bb = b2[n];
      for (int r = 0; r < 8; ++r) {
        int R = rowBase + r + hi * 8;
        Cout[R * 1024 + n] = (f16)tanhf(a2[f][r] + bb);
      }
    }
  }
}

// ---------------------------------------------------------------------------
// Kernel 3: oscillator simulation. One WG per 8 batches (N = 16 cols = one
// WMMA N-tile). x state in owner-thread registers + LDS f16 B-fragment
// buffer; each inner step: y = cs * (J @ x) via WMMA, then elementwise
// update + per-(b,j) 2-vector normalization.
// ---------------------------------------------------------------------------
__global__ __launch_bounds__(256) void sim_kernel(
    const f16* __restrict__ Jp,     // packed J fragments
    const f16* __restrict__ Cbuf,   // 32768 x 1024 f16
    const float* __restrict__ omega,
    const float* __restrict__ csp,  // coupling_strength scalar
    float* __restrict__ out)        // (256,128,512,2) f32
{
  __shared__ __align__(16) f16   sX[16 * 520];   // x, col-major [col][j], stride 520
  __shared__ __align__(16) float sY[16 * 516];   // y, col-major [col][m], stride 516

  const int tid = threadIdx.x;
  const int wv  = tid >> 5;
  const int ln  = tid & 31;
  const int lc  = ln & 15;
  const int hi  = ln >> 4;
  const int wg  = blockIdx.x;      // batches wg*8 .. wg*8+7
  const float cs = csp[0];

  // Thread owns 16 (bl, j) pairs: bl = k>>1, j = tid + 256*(k&1).
  float x0[16], x1[16], xi0[16], xi1[16], om[16];
  for (int k = 0; k < 16; ++k) {
    int bl = k >> 1;
    int j  = tid + ((k & 1) << 8);
    om[k] = omega[j];
    float ang = 6.283185307179586f * (float)j * (1.0f / 512.0f);
    float c = cosf(ang), s = sinf(ang);
    x0[k] = c; x1[k] = s;
    sX[(bl * 2) * 520 + j]     = (f16)c;
    sX[(bl * 2 + 1) * 520 + j] = (f16)s;
  }

  for (int t = 0; t < 128; ++t) {
    float c0[16], c1[16];
    for (int k = 0; k < 16; ++k) {
      int bl = k >> 1;
      int j  = tid + ((k & 1) << 8);
      int R = (wg * 8 + bl) * 128 + t;
      union { unsigned u; f16 h[2]; } cu;
      cu.u = *(const unsigned*)(Cbuf + R * 1024 + j * 2);
      c0[k] = (float)cu.h[0];
      c1[k] = (float)cu.h[1];
      xi0[k] = x0[k]; xi1[k] = x1[k];
    }
    for (int s = 0; s < 6; ++s) {
      __syncthreads();             // sX ready
      // GEMM: wave wv handles m-tiles wv*4 .. wv*4+3
      for (int q = 0; q < 4; ++q) {
        int mt = wv * 4 + q;
        v8f acc = zero8();
        for (int kt = 0; kt < 16; ++kt) {
          const f16* ap = Jp + (((mt * 16 + kt) * 32) + ln) * 16;
          if (kt + 1 < 16)  // hide L2 latency on next fragment
            __builtin_prefetch(Jp + (((mt * 16 + kt + 1) * 32) + ln) * 16, 0, 1);
          v16h a = ld_frag2(ap, ap + 8);
          const f16* bp = &sX[lc * 520 + kt * 32 + hi * 8];
          v16h b = ld_frag2(bp, bp + 16);
          acc = __builtin_amdgcn_wmma_f32_16x16x32_f16(
              false, a, false, b, (short)0, acc, false, false);
        }
        float4* yp = (float4*)&sY[lc * 516 + mt * 16 + hi * 8];
        yp[0] = make_float4(acc[0] * cs, acc[1] * cs, acc[2] * cs, acc[3] * cs);
        yp[1] = make_float4(acc[4] * cs, acc[5] * cs, acc[6] * cs, acc[7] * cs);
      }
      __syncthreads();             // sY ready
      // Elementwise update + normalize
      for (int k = 0; k < 16; ++k) {
        int bl = k >> 1;
        int j  = tid + ((k & 1) << 8);
        float y0 = sY[(bl * 2) * 516 + j];
        float y1 = sY[(bl * 2 + 1) * 516 + j];
        float nx0 = x0[k] + DT_C * (-om[k] * x1[k] + c0[k] + y0);
        float nx1 = x1[k] + DT_C * ( om[k] * x0[k] + c1[k] + y1);
        float nrm = sqrtf(nx0 * nx0 + nx1 * nx1);
        float inv = 1.0f / fmaxf(nrm, 1e-12f);
        nx0 *= inv; nx1 *= inv;
        x0[k] = nx0; x1[k] = nx1;
        sX[(bl * 2) * 520 + j]     = (f16)nx0;
        sX[(bl * 2 + 1) * 520 + j] = (f16)nx1;
      }
    }
    // Combine with x_init, normalize, emit, carry forward.
    for (int k = 0; k < 16; ++k) {
      int bl = k >> 1;
      int j  = tid + ((k & 1) << 8);
      float fx0 = 0.9f * x0[k] + 0.1f * xi0[k];
      float fx1 = 0.9f * x1[k] + 0.1f * xi1[k];
      float nrm = sqrtf(fx0 * fx0 + fx1 * fx1);
      float inv = 1.0f / fmaxf(nrm, 1e-12f);
      fx0 *= inv; fx1 *= inv;
      x0[k] = fx0; x1[k] = fx1;
      sX[(bl * 2) * 520 + j]     = (f16)fx0;
      sX[(bl * 2 + 1) * 520 + j] = (f16)fx1;
      int R = (wg * 8 + bl) * 128 + t;
      ((float2*)out)[R * 512 + j] = make_float2(fx0, fx1);
    }
  }
}

extern "C" void kernel_launch(void* const* d_in, const int* in_sizes, int n_in,
                              void* d_out, int out_size, void* d_ws, size_t ws_size,
                              hipStream_t stream) {
  const float* feat  = (const float*)d_in[0];
  const float* omega = (const float*)d_in[1];
  const float* J     = (const float*)d_in[2];
  const float* W1    = (const float*)d_in[3];
  const float* b1    = (const float*)d_in[4];
  const float* lng   = (const float*)d_in[5];
  const float* lnb   = (const float*)d_in[6];
  const float* W2    = (const float*)d_in[7];
  const float* b2    = (const float*)d_in[8];
  const float* cs    = (const float*)d_in[9];
  float* out = (float*)d_out;

  // Workspace layout: [0, 512KB) packed J (f16), then 64MB c_all (f16).
  f16* Jp   = (f16*)d_ws;
  f16* Cbuf = (f16*)((char*)d_ws + 512 * 1024);

  packJ_kernel<<<64, 256, 0, stream>>>(J, Jp);
  mlp_kernel<<<2048, 256, 0, stream>>>(feat, W1, b1, lng, lnb, W2, b2, Cbuf);
  sim_kernel<<<32, 256, 0, stream>>>(Jp, Cbuf, omega, cs, out);
}
